// AttnBlock_73804718014778
// MI455X (gfx1250) — matrix operand order repaired and checked by reading, not compiled, not run on hardware
//
#include <hip/hip_runtime.h>

typedef __bf16 bf16;
typedef __attribute__((ext_vector_type(16))) __bf16 v16bf;
typedef __attribute__((ext_vector_type(8)))  __bf16 v8bf;
typedef __attribute__((ext_vector_type(8)))  float  v8f;
typedef __attribute__((ext_vector_type(4))) unsigned int v4u;
typedef __attribute__((ext_vector_type(8))) int v8i_t;
typedef __attribute__((ext_vector_type(4))) int v4i_t;

static constexpr int TT = 32768;   // T
static constexpr float GN_EPS = 1e-5f;

#define WMMA_BF16(A, B, C) \
  __builtin_amdgcn_wmma_f32_16x16x32_bf16(false, (A), false, (B), (short)0, (C), false, false)
#define CAT16(lo, hi) \
  __builtin_shufflevector((lo), (hi), 0, 1, 2, 3, 4, 5, 6, 7, 8, 9, 10, 11, 12, 13, 14, 15)

// ---- TDM: 2D tile (128x128, 2-byte elems) LDS <-> global -------------------
// D# per CDNA5 ISA §8.3/8.4: group0 = {count=1, lds_addr, global_addr, type=2},
// group1 = {data_size=2B, tensor_dim0/1, tile_dim0/1, tensor_dim0_stride}.
// 6-arg builtin form (clang-23 / therock headers): extra int32x8 group + cpol.
__device__ __forceinline__ void tdm_tile_2d(unsigned lds_addr,
                                            const void* gaddr,
                                            unsigned long long stride0_elems,
                                            bool store) {
  const unsigned long long ga = (unsigned long long)gaddr;
  v4u g0;
  g0[0] = 1u;                                            // count=1 (user D#)
  g0[1] = lds_addr;                                      // LDS byte address
  g0[2] = (unsigned)ga;                                  // global_addr[31:0]
  g0[3] = (unsigned)((ga >> 32) & 0x01ffffffu) | (2u << 30);  // [56:32]|type=2
  v8i_t g1;
  g1[0] = (int)(1u << 16);                        // data_size=1 -> 2 bytes
  g1[1] = (int)((32768u & 0xffffu) << 16);        // tensor_dim0 lo16 << 16
  g1[2] = (int)((32768u >> 16) | (128u << 16));   // dim0 hi16 | tensor_dim1 lo
  g1[3] = (int)(128u << 16);                      // dim1 hi | tile_dim0=128
  g1[4] = 128;                                    // tile_dim1=128, tile_dim2=0
  g1[5] = (int)(stride0_elems & 0xffffffffu);     // tensor_dim0_stride lo32
  g1[6] = (int)((stride0_elems >> 32) & 0xffffu); // stride0 hi16 | stride1 lo
  g1[7] = 0;
  v4i_t z4;
  z4[0] = z4[1] = z4[2] = z4[3] = 0;
  v8i_t z8;
  z8[0] = z8[1] = z8[2] = z8[3] = z8[4] = z8[5] = z8[6] = z8[7] = 0;
  if (store)
    __builtin_amdgcn_tensor_store_from_lds(g0, g1, z4, z4, z8, 0);
  else
    __builtin_amdgcn_tensor_load_to_lds(g0, g1, z4, z4, z8, 0);
}

// ---- WMMA operand loaders (CDNA5 ISA 7.12.2, wave32) -----------------------
__device__ __forceinline__ v16bf load_a16(const bf16* p, int ld) {
  const int lane = threadIdx.x & 31;
  const bf16* r = p + (lane & 15) * ld + ((lane >> 4) << 4);
  const v8bf lo = *(const v8bf*)(r);
  const v8bf hi = *(const v8bf*)(r + 8);
  return CAT16(lo, hi);
}

__device__ __forceinline__ v16bf load_bT(const bf16* p, int ld) {
  const int lane = threadIdx.x & 31;
  const bf16* r = p + (lane & 15) * ld + ((lane >> 4) << 4);
  const v8bf lo = *(const v8bf*)(r);
  const v8bf hi = *(const v8bf*)(r + 8);
  return CAT16(lo, hi);
}

// B from XOR-swizzled transposed buffer (128 K per row, 8-elem groups):
// element (n,k) at n*128 + (((k>>3)^(n&15))<<3) + (k&7)
__device__ __forceinline__ v16bf load_bT_sw(const bf16* buf, int n0, int koff) {
  const int lane = threadIdx.x & 31;
  const int nm = lane & 15;
  const int g0 = (koff >> 3) + ((lane >> 4) << 1);
  const bf16* row = buf + (n0 + nm) * 128;
  const v8bf lo = *(const v8bf*)(row + ((g0 ^ nm) << 3));
  const v8bf hi = *(const v8bf*)(row + (((g0 + 1) ^ nm) << 3));
  return CAT16(lo, hi);
}

__device__ __forceinline__ int sw_addr(int d, int i) {  // swizzled (d,i) offset
  return d * 128 + (((i >> 3) ^ (d & 15)) << 3) + (i & 7);
}

// ---- K0: zero gram accumulator, convert 4 weight matrices to bf16 ----------
__global__ __launch_bounds__(256) void k0_prep(
    float* __restrict__ gram, const float* __restrict__ wq,
    const float* __restrict__ wk, const float* __restrict__ wv,
    const float* __restrict__ wp, bf16* __restrict__ wbf) {
  const int id = blockIdx.x * 256 + threadIdx.x;
  if (id < 8 * 128 * 128) gram[id] = 0.0f;
  if (id < 4 * 128 * 128) {
    const int m = id >> 14, e = id & 16383;
    const float* src = (m == 0) ? wq : (m == 1) ? wk : (m == 2) ? wv : wp;
    wbf[id] = (bf16)src[e];
  }
}

// ---- K1: GroupNorm statistics. One block per (b, g): 4 channels x T --------
__global__ __launch_bounds__(256) void k1_stats(const float* __restrict__ x,
                                                float* __restrict__ stats) {
  __shared__ float s0[256], s1[256];
  const int tid = threadIdx.x;
  const float* p = x + (size_t)blockIdx.x * 4 * TT;  // (b*32+g)*4*T
  float sum = 0.f, sq = 0.f;
#pragma unroll 4
  for (int s = 0; s < 128; ++s) {
    const float4 v = *(const float4*)(p + (((size_t)tid + (s << 8)) << 2));
    sum += v.x + v.y + v.z + v.w;
    sq += v.x * v.x + v.y * v.y + v.z * v.z + v.w * v.w;
  }
  s0[tid] = sum; s1[tid] = sq;
  __syncthreads();
  for (int st = 128; st > 0; st >>= 1) {
    if (tid < st) { s0[tid] += s0[tid + st]; s1[tid] += s1[tid + st]; }
    __syncthreads();
  }
  if (tid == 0) {
    const float n = 4.0f * TT;
    const float mean = s0[0] / n;
    const float var = s1[0] / n - mean * mean;
    stats[blockIdx.x * 2] = mean;
    stats[blockIdx.x * 2 + 1] = rsqrtf(var + GN_EPS);
  }
}

// ---- K2: fused GN-apply + Q/K/V projections + Gram accumulation + V spill --
__global__ __launch_bounds__(256) void k2_qkv_gram(
    const float* __restrict__ x, const float* __restrict__ gamma,
    const float* __restrict__ beta, const float* __restrict__ stats,
    const bf16* __restrict__ wq, const bf16* __restrict__ wk,
    const bf16* __restrict__ wv, const float* __restrict__ bq,
    const float* __restrict__ bk, const float* __restrict__ bv,
    float* __restrict__ gram, bf16* __restrict__ vout) {
  __shared__ __align__(16) bf16 sh_h[128 * 128];  // swizzled hT / v staging
  __shared__ __align__(16) bf16 sh_q[128 * 40];   // qT quarter [c][il]
  __shared__ __align__(16) bf16 sh_k[128 * 40];   // kT quarter [d][il]

  const int tid = threadIdx.x;
  const int lane = tid & 31;
  const int wid = tid >> 5;
  const int lh = lane >> 4;
  const int ln = lane & 15;
  const int o0 = wid * 16;  // this wave's output-row strip (also gram c-strip)

  const int b = blockIdx.x >> 6;
  const int jg = blockIdx.x & 63;
  const unsigned lds_stage = (unsigned)(size_t)(void*)sh_h;

  v8f gacc[8];
#pragma unroll
  for (int nt = 0; nt < 8; ++nt)
#pragma unroll
    for (int e = 0; e < 8; ++e) gacc[nt][e] = 0.0f;

  float bqe[8], bke[8], bve[8];
#pragma unroll
  for (int e = 0; e < 8; ++e) {
    const int o = o0 + e + 8 * lh;
    bqe[e] = bq[o]; bke[e] = bk[o]; bve[e] = bv[o];
  }

  for (int jt = 0; jt < 4; ++jt) {
    const int t0 = (((jg << 2) + jt) << 7);
    // previous iteration's TDM store must finish reading sh_h first
    if (wid == 0) __builtin_amdgcn_s_wait_tensorcnt(0);
    __syncthreads();
    // load + normalize x tile -> swizzled transposed sh_h[d][i]
#pragma unroll
    for (int s = 0; s < 16; ++s) {
      const int idx = tid + (s << 8);
      const int i = idx >> 5, c4 = idx & 31;
      const float4 xv =
          *(const float4*)(x + (size_t)(b * 128 + i) * TT + t0 + (c4 << 2));
      const int g = ((b << 5) + (i >> 2)) << 1;
      const float rstd = stats[g + 1];
      const float sc = rstd * gamma[i];
      const float sf = beta[i] - stats[g] * sc;
      const int d0 = c4 << 2;
      sh_h[sw_addr(d0 + 0, i)] = (bf16)(xv.x * sc + sf);
      sh_h[sw_addr(d0 + 1, i)] = (bf16)(xv.y * sc + sf);
      sh_h[sw_addr(d0 + 2, i)] = (bf16)(xv.z * sc + sf);
      sh_h[sw_addr(d0 + 3, i)] = (bf16)(xv.w * sc + sf);
    }
    __syncthreads();

    // Q and K projections (share B-operand loads)
    v8f qa[8], ka[8];
#pragma unroll
    for (int nt = 0; nt < 8; ++nt)
#pragma unroll
      for (int e = 0; e < 8; ++e) { qa[nt][e] = bqe[e]; ka[nt][e] = bke[e]; }
#pragma unroll
    for (int kk = 0; kk < 4; ++kk) {
      const v16bf aq = load_a16(wq + o0 * 128 + kk * 32, 128);
      const v16bf ak = load_a16(wk + o0 * 128 + kk * 32, 128);
#pragma unroll
      for (int nt = 0; nt < 8; ++nt) {
        const v16bf bm = load_bT_sw(sh_h, nt * 16, kk * 32);
        qa[nt] = WMMA_BF16(aq, bm, qa[nt]);
        ka[nt] = WMMA_BF16(ak, bm, ka[nt]);
      }
    }

    // Gram += qT @ k over K=i in four 32-channel quarter-phases
#pragma unroll
    for (int ph = 0; ph < 4; ++ph) {
      __syncthreads();
      if ((wid >> 1) == ph) {
        const int ib = o0 - ph * 32;  // local i-strip base: 0 or 16
#pragma unroll
        for (int nt = 0; nt < 8; ++nt)
#pragma unroll
          for (int e = 0; e < 8; ++e) {
            const int il = ib + e + 8 * lh;
            const int cc = nt * 16 + ln;
            sh_q[cc * 40 + il] = (bf16)qa[nt][e];
            sh_k[cc * 40 + il] = (bf16)ka[nt][e];
          }
      }
      __syncthreads();
      const v16bf ag = load_a16(sh_q + o0 * 40, 40);
#pragma unroll
      for (int nt = 0; nt < 8; ++nt) {
        const v16bf bg = load_bT(sh_k + (nt * 16) * 40, 40);
        gacc[nt] = WMMA_BF16(ag, bg, gacc[nt]);
      }
    }

    // V projection (sh_h still intact)
    v8f va[8];
#pragma unroll
    for (int nt = 0; nt < 8; ++nt)
#pragma unroll
      for (int e = 0; e < 8; ++e) va[nt][e] = bve[e];
#pragma unroll
    for (int kk = 0; kk < 4; ++kk) {
      const v16bf av = load_a16(wv + o0 * 128 + kk * 32, 128);
#pragma unroll
      for (int nt = 0; nt < 8; ++nt) {
        const v16bf bm = load_bT_sw(sh_h, nt * 16, kk * 32);
        va[nt] = WMMA_BF16(av, bm, va[nt]);
      }
    }
    __syncthreads();
    // stage v tile natural [i][d] into sh_h, then spill via TDM (DMA engine)
#pragma unroll
    for (int nt = 0; nt < 8; ++nt)
#pragma unroll
      for (int e = 0; e < 8; ++e)
        sh_h[(o0 + e + 8 * lh) * 128 + nt * 16 + ln] = (bf16)va[nt][e];
    __syncthreads();
    if (wid == 0)
      tdm_tile_2d(lds_stage, vout + (size_t)(b * 128) * TT + t0,
                  (unsigned long long)TT, /*store=*/true);
  }

  // flush this block's partial Gram
  float* gb = gram + b * 16384;
#pragma unroll
  for (int nt = 0; nt < 8; ++nt)
#pragma unroll
    for (int e = 0; e < 8; ++e)
      atomicAdd(&gb[(o0 + e + 8 * lh) * 128 + nt * 16 + ln], gacc[nt][e]);
}

// ---- K3: scale + row softmax; rows written natural ([d][c] == B layout) ----
__global__ void k3_softmax(const float* __restrict__ gram,
                           bf16* __restrict__ wt) {
  const int b = blockIdx.x, r = threadIdx.x;  // 128 threads: one gram row each
  const float* row = gram + b * 16384 + r * 128;
  const float sc = 0.005524271728019903f;  // 32768^-0.5
  float m = -3.4e38f;
  for (int c = 0; c < 128; ++c) m = fmaxf(m, row[c] * sc);
  float s = 0.f;
  for (int c = 0; c < 128; ++c) s += __expf(row[c] * sc - m);
  const float inv = 1.0f / s;
  bf16* dst = wt + b * 16384 + r * 128;
  for (int c = 0; c < 128; ++c) dst[c] = (bf16)(__expf(row[c] * sc - m) * inv);
}

// ---- K4: fused (V @ Wt) -> (Wp @ hatt) + bias + residual -------------------
__global__ __launch_bounds__(256) void k4_apply(
    const bf16* __restrict__ vin, const bf16* __restrict__ wt,
    const bf16* __restrict__ wp, const float* __restrict__ bp,
    const float* __restrict__ x, float* __restrict__ out) {
  __shared__ __align__(16) bf16 sh_v[128 * 128];  // natural [i][c] (A operand)
  __shared__ __align__(16) bf16 sh_a[128 * 128];  // swizzled hattT [d][i]
  const int tid = threadIdx.x;
  const int lane = tid & 31;
  const int wid = tid >> 5;
  const int lh = lane >> 4;
  const int ln = lane & 15;
  const int o0 = wid * 16;
  const int b = blockIdx.x >> 8;
  const int t0 = (blockIdx.x & 255) << 7;

  // v tile: one TDM DMA instead of 2048 load/store pairs
  if (wid == 0) {
    tdm_tile_2d((unsigned)(size_t)(void*)sh_v,
                vin + (size_t)(b * 128) * TT + t0, (unsigned long long)TT,
                /*store=*/false);
    __builtin_amdgcn_s_wait_tensorcnt(0);
  }
  __syncthreads();

  // GEMM1: hatt[i,d] = sum_c v[i,c] * w_sm[d,c]; B = wt rows (global, [d][c])
  const bf16* wtb = wt + b * 16384;
  v8f acc[8];
#pragma unroll
  for (int nt = 0; nt < 8; ++nt)
#pragma unroll
    for (int e = 0; e < 8; ++e) acc[nt][e] = 0.0f;
#pragma unroll
  for (int kk = 0; kk < 4; ++kk) {
    const v16bf a = load_a16(sh_v + o0 * 128 + kk * 32, 128);
#pragma unroll
    for (int nt = 0; nt < 8; ++nt) {
      const v16bf bm = load_bT(wtb + (nt * 16) * 128 + kk * 32, 128);
      acc[nt] = WMMA_BF16(a, bm, acc[nt]);
    }
  }
  // stage hatt transposed+swizzled [d][i] (conflict-free scattered stores)
#pragma unroll
  for (int nt = 0; nt < 8; ++nt)
#pragma unroll
    for (int e = 0; e < 8; ++e)
      sh_a[sw_addr(nt * 16 + ln, o0 + e + 8 * lh)] = (bf16)acc[nt][e];
  __syncthreads();

  // GEMM2: out[o,d] = bp[o] + sum_i wp[o,i] * hatt[i,d] + x[o,d]
  v8f po[8];
#pragma unroll
  for (int e = 0; e < 8; ++e) {
    const float bpv = bp[o0 + e + 8 * lh];
#pragma unroll
    for (int nt = 0; nt < 8; ++nt) po[nt][e] = bpv;
  }
#pragma unroll
  for (int kk = 0; kk < 4; ++kk) {
    const v16bf a = load_a16(wp + o0 * 128 + kk * 32, 128);
#pragma unroll
    for (int nt = 0; nt < 8; ++nt) {
      const v16bf bm = load_bT_sw(sh_a, nt * 16, kk * 32);
      po[nt] = WMMA_BF16(a, bm, po[nt]);
    }
  }
#pragma unroll
  for (int nt = 0; nt < 8; ++nt)
#pragma unroll
    for (int e = 0; e < 8; ++e) {
      const int o = o0 + e + 8 * lh;
      const size_t off = (size_t)(b * 128 + o) * TT + t0 + nt * 16 + ln;
      out[off] = po[nt][e] + x[off];
    }
}

extern "C" void kernel_launch(void* const* d_in, const int* in_sizes, int n_in,
                              void* d_out, int out_size, void* d_ws,
                              size_t ws_size, hipStream_t stream) {
  (void)in_sizes; (void)n_in; (void)out_size; (void)ws_size;
  const float* x    = (const float*)d_in[0];
  const float* gn_w = (const float*)d_in[1];
  const float* gn_b = (const float*)d_in[2];
  const float* wq   = (const float*)d_in[3];
  const float* bq   = (const float*)d_in[4];
  const float* wk   = (const float*)d_in[5];
  const float* bk   = (const float*)d_in[6];
  const float* wv   = (const float*)d_in[7];
  const float* bv   = (const float*)d_in[8];
  const float* wp   = (const float*)d_in[9];
  const float* bp   = (const float*)d_in[10];
  float* out = (float*)d_out;

  char* ws = (char*)d_ws;
  float* gram  = (float*)(ws);                 // 8*128*128 f32  = 512 KB
  float* stats = (float*)(ws + 524288);        // 256*2 f32      = 2 KB
  bf16*  wbf   = (bf16*)(ws + 526336);         // 4*16384 bf16   = 128 KB
  bf16*  wtb   = (bf16*)(ws + 657408);         // 8*16384 bf16   = 256 KB
  bf16*  vbuf  = (bf16*)(ws + 1048576);        // 8*128*32768    = 64 MB

  k0_prep<<<512, 256, 0, stream>>>(gram, wq, wk, wv, wp, wbf);
  k1_stats<<<256, 256, 0, stream>>>(x, stats);
  k2_qkv_gram<<<512, 256, 0, stream>>>(x, gn_w, gn_b, stats, wbf, wbf + 16384,
                                       wbf + 32768, bq, bk, bv, gram, vbuf);
  k3_softmax<<<8, 128, 0, stream>>>(gram, wtb);
  k4_apply<<<2048, 256, 0, stream>>>(vbuf, wtb, wbf + 49152, bp, x, out);
}
